// BiLSTM_CRF_10531259810720
// MI455X (gfx1250) — matrix operand order repaired and checked by reading, not compiled, not run on hardware
//
#include <hip/hip_runtime.h>

// ---------------------------------------------------------------------------
// BiLSTM-CRF for MI455X (gfx1250, wave32, WMMA f16 16x16x32 -> f32 accum)
// GEMM row space is t-major (m = t*32 + b) so GEMM C-tiles == LSTM batch-tiles,
// letting xg live in C-fragment tile layout ([tile][lane][8] f16, b128 I/O).
// ---------------------------------------------------------------------------

typedef _Float16 half_t;
typedef __attribute__((ext_vector_type(16))) _Float16 v16h;
typedef __attribute__((ext_vector_type(8)))  _Float16 v8h;
typedef __attribute__((ext_vector_type(8)))  float    v8f;

#define B_   32
#define T_   512
#define D_   300
#define DP_  320          // D padded to K multiple of 32
#define HID_ 512
#define H2_  256
#define NT_  9
#define M_   (B_ * T_)    // 16384 rows

__device__ __forceinline__ v8f wmma16(v16h a, v16h b, v8f c) {
  return __builtin_amdgcn_wmma_f32_16x16x32_f16(false, a, false, b, (short)0, c,
                                                false, false);
}
__device__ __forceinline__ float sigm(float x) { return 1.0f / (1.0f + __expf(-x)); }

// Fragment layouts (ISA 7.12.2, wave32):
//  A frag (16x32): lane L: row m = L&15 ; half j -> k = (L>>4)*8 + (j<8 ? j : j+8)
//  B frag (32x16): lane L: col n = L&15 ; half j -> k = (L>>4)*8 + (j<8 ? j : j+8)
//  C tile (16x16): lane L, elem v: n = L&15 ; m = (L>>4)*8 + v
// Packed frags: 512 halfs each, lane-contiguous 16 halfs (32B).

// ---- lens / rev indices -----------------------------------------------------
__global__ void k_prep(const int* __restrict__ word, int* __restrict__ rev,
                       float* __restrict__ outLens) {
  int b = blockIdx.x;
  int t = threadIdx.x;
  __shared__ int red[T_];
  red[t] = (word[b * T_ + t] > 0) ? 1 : 0;
  __syncthreads();
  for (int s = T_ / 2; s > 0; s >>= 1) {
    if (t < s) red[t] += red[t + s];
    __syncthreads();
  }
  int len = red[0];
  rev[b * T_ + t] = (t < len) ? (len - 1 - t) : t;
  if (t == 0) outLens[b] = (float)len;
}

// ---- combined biases --------------------------------------------------------
__global__ void k_bias(const float* bih0, const float* bhh0,
                       const float* bih1, const float* bhh1,
                       const float* bout,
                       float* b0, float* b1, float* bo) {
  int i = blockIdx.x * blockDim.x + threadIdx.x;
  if (i < 2048) { b0[i] = bih0[i] + bhh0[i]; b1[i] = bih1[i] + bhh1[i]; }
  if (i < 32)   { bo[i] = (i < NT_) ? bout[i] : 0.0f; }
}

// ---- pack weight [N,K] f32 row-major -> B-fragments f16 ---------------------
__global__ void k_pack_B(const float* __restrict__ W, half_t* __restrict__ out,
                         int N, int K, int Kf, int nfrag) {
  int idx  = blockIdx.x * blockDim.x + threadIdx.x;
  int lane = idx & 31;
  int frag = idx >> 5;
  if (frag >= nfrag) return;
  int kf = frag % Kf, nt = frag / Kf;
  int n = nt * 16 + (lane & 15);
  int kbase = kf * 32 + (lane >> 4) * 8;
  v16h tv;
#pragma unroll
  for (int j = 0; j < 16; ++j) {
    int k = kbase + (j < 8 ? j : j + 8);
    tv[j] = (half_t)((n < N && k < K) ? W[(size_t)n * K + k] : 0.0f);
  }
  *(v16h*)(out + (size_t)frag * 512 + lane * 16) = tv;
}

// ---- pack A from embedding gather (layer 0, optional reverse) ---------------
// GEMM rows are t-major: m = t*32 + b.
__global__ void k_pack_A_embed(const int* __restrict__ word, const int* __restrict__ rev,
                               const float* __restrict__ embed, half_t* __restrict__ out,
                               int useRev, int nfrag) {
  int idx  = blockIdx.x * blockDim.x + threadIdx.x;
  int lane = idx & 31;
  int frag = idx >> 5;
  if (frag >= nfrag) return;
  const int Kf = DP_ / 32;                       // 10
  int kf = frag % Kf, mt = frag / Kf;
  int m = mt * 16 + (lane & 15);
  int b = m & (B_ - 1), t = m >> 5;
  int ts = useRev ? rev[b * T_ + t] : t;
  const float* src = embed + (size_t)word[b * T_ + ts] * D_;
  int g2 = lane >> 4;
  v16h tv;
#pragma unroll
  for (int j = 0; j < 16; ++j) {
    int k = kf * 32 + g2 * 8 + (j < 8 ? j : j + 8);
    tv[j] = (half_t)(k < D_ ? src[k] : 0.0f);
  }
  *(v16h*)(out + (size_t)frag * 512 + lane * 16) = tv;
}

// ---- pack A from f16 activations x[(b*T+t), C] (optional reverse) -----------
__global__ void k_pack_A_h(const half_t* __restrict__ x, const int* __restrict__ rev,
                           half_t* __restrict__ out, int C, int useRev, int nfrag) {
  int idx  = blockIdx.x * blockDim.x + threadIdx.x;
  int lane = idx & 31;
  int frag = idx >> 5;
  if (frag >= nfrag) return;
  int Kf = C / 32;
  int kf = frag % Kf, mt = frag / Kf;
  int m = mt * 16 + (lane & 15);
  int b = m & (B_ - 1), t = m >> 5;
  int ts = useRev ? rev[b * T_ + t] : t;
  const half_t* src = x + (size_t)(b * T_ + ts) * C;
  int g2 = lane >> 4;
  v16h tv;
#pragma unroll
  for (int j = 0; j < 16; ++j) {
    int k = kf * 32 + g2 * 8 + (j < 8 ? j : j + 8);
    tv[j] = src[k];
  }
  *(v16h*)(out + (size_t)frag * 512 + lane * 16) = tv;
}

// ---- WMMA GEMM: 2x2 tiles per wave, 4 wmma per K-step -----------------------
// mode 0: f32 row-major [m, ldc] + bias (emissions)
// mode 2: f16 C-tile layout [(mt*Nt+nt)*256 + lane*8] + bias (xg handoff)
__global__ void k_gemm(const half_t* __restrict__ A, const half_t* __restrict__ Bp,
                       const float* __restrict__ bias, void* __restrict__ Cout,
                       int Nt, int Kf, int ldc, int mode) {
  int wid  = (blockIdx.x * blockDim.x + threadIdx.x) >> 5;
  int lane = threadIdx.x & 31;
  int nw = Nt >> 1;
  int mt0 = (wid / nw) * 2;
  int nt0 = (wid % nw) * 2;
  v8f zero = {};
  v8f c00 = zero, c01 = zero, c10 = zero, c11 = zero;
  const half_t* a0p = A  + ((size_t)mt0 * Kf) * 512 + lane * 16;
  const half_t* a1p = a0p + (size_t)Kf * 512;
  const half_t* b0p = Bp + ((size_t)nt0 * Kf) * 512 + lane * 16;
  const half_t* b1p = b0p + (size_t)Kf * 512;
  for (int kf = 0; kf < Kf; ++kf) {
    v16h a0 = *(const v16h*)a0p;
    v16h a1 = *(const v16h*)a1p;
    v16h b0 = *(const v16h*)b0p;
    v16h b1 = *(const v16h*)b1p;
    // prefetch 4 K-steps ahead (matches x4 unroll), near scope
    __builtin_prefetch(a0p + 4 * 512, 0, 3);
    __builtin_prefetch(a1p + 4 * 512, 0, 3);
    __builtin_prefetch(b0p + 4 * 512, 0, 3);
    __builtin_prefetch(b1p + 4 * 512, 0, 3);
    a0p += 512; a1p += 512; b0p += 512; b1p += 512;
    c00 = wmma16(a0, b0, c00);
    c01 = wmma16(a0, b1, c01);
    c10 = wmma16(a1, b0, c10);
    c11 = wmma16(a1, b1, c11);
  }
  int lrow = lane >> 4, lcol = lane & 15;
  v8f acc[2][2] = {{c00, c01}, {c10, c11}};
#pragma unroll
  for (int i = 0; i < 2; ++i)
#pragma unroll
    for (int j = 0; j < 2; ++j) {
      int nt = nt0 + j;
      float bv = bias ? bias[nt * 16 + lcol] : 0.0f;
      if (mode == 2) {
        v8h hv;
#pragma unroll
        for (int v = 0; v < 8; ++v) hv[v] = (half_t)(acc[i][j][v] + bv);
        *(v8h*)((half_t*)Cout + ((size_t)(mt0 + i) * Nt + nt) * 256 + lane * 8) = hv;
      } else {
#pragma unroll
        for (int v = 0; v < 8; ++v) {
          int m = (mt0 + i) * 16 + lrow * 8 + v;
          ((float*)Cout)[(size_t)m * ldc + nt * 16 + lcol] = acc[i][j][v] + bv;
        }
      }
    }
}

// ---- persistent BiLSTM recurrence: one block per direction ------------------
// Block = 1024 threads = 32 waves = (batch-tile mt, h-col-tile jt). Each wave
// owns all four gate tiles (i,f,g,o) of its (mt,jt): the lane mapping of the
// four C tiles is identical, so the cell update is register-resident per lane.
// xg is in C-tile layout: tile (t*2+mt, g*16+jt) -> one b128 load per gate.
__global__ __launch_bounds__(1024) void k_lstm(
    const half_t* __restrict__ xgAll,   // [2][1024*64*256] f16 tile layout
    const half_t* __restrict__ BhhAll,  // [2][64*8*512] packed W_hh frags
    const int* __restrict__ word, const int* __restrict__ rev,
    half_t* __restrict__ xout)          // [(b*T+t), 512]: 0..255 fwd, 256..511 bwd
{
  int dir = blockIdx.x;
  const half_t* xg  = xgAll  + (size_t)dir * 1024 * 64 * 256;
  const half_t* Bhh = BhhAll + (size_t)dir * 64 * 8 * 512;
  int tid  = threadIdx.x;
  int lane = tid & 31;
  int w    = tid >> 5;
  int mt   = w >> 4;              // 0..1  (batch tile)
  int jt   = w & 15;              // 0..15 (h-column tile)
  int lrow = lane >> 4, lcol = lane & 15;
  int jcol = jt * 16 + lcol;      // h column 0..255

  __shared__ __align__(32) half_t hA[2 * 8 * 512];   // h in A-frag layout, 16KB

  for (int i = tid; i < 2 * 8 * 512; i += 1024) hA[i] = (half_t)0;
  __syncthreads();

  float c[8];
#pragma unroll
  for (int v = 0; v < 8; ++v) c[v] = 0.0f;

  // hA write slot for this lane's (b, jcol): invert the A-frag k-mapping
  int kf_w = jcol >> 5;
  int kk   = jcol & 31;
  int g2w  = (kk & 15) >> 3;
  int jjw  = (kk & 7) + ((kk >> 4) << 3);
  int baseW = ((mt * 8 + kf_w) * 32 + g2w * 16 + lrow * 8) * 16 + jjw;

  const half_t* Bg0 = Bhh + ((size_t)((0 * 16 + jt) * 8)) * 512 + lane * 16;
  const half_t* Bg1 = Bhh + ((size_t)((1 * 16 + jt) * 8)) * 512 + lane * 16;
  const half_t* Bg2 = Bhh + ((size_t)((2 * 16 + jt) * 8)) * 512 + lane * 16;
  const half_t* Bg3 = Bhh + ((size_t)((3 * 16 + jt) * 8)) * 512 + lane * 16;
  const half_t* hAbase = hA + (size_t)(mt * 8) * 512 + lane * 16;

  for (int t = 0; t < T_; ++t) {
    // gate init: 4 contiguous 16B loads from tile-layout xg
    const half_t* xgt = xg + ((size_t)(t * 2 + mt) * 64) * 256 + (size_t)jt * 256 + lane * 8;
    v8h qi = *(const v8h*)(xgt + 0 * 16 * 256);
    v8h qf = *(const v8h*)(xgt + 1 * 16 * 256);
    v8h qg = *(const v8h*)(xgt + 2 * 16 * 256);
    v8h qo = *(const v8h*)(xgt + 3 * 16 * 256);
    v8f ci, cf, cg, co;
#pragma unroll
    for (int v = 0; v < 8; ++v) {
      ci[v] = (float)qi[v];
      cf[v] = (float)qf[v];
      cg[v] = (float)qg[v];
      co[v] = (float)qo[v];
    }
#pragma unroll
    for (int kf = 0; kf < 8; ++kf) {
      v16h a  = *(const v16h*)(hAbase + (size_t)kf * 512);
      v16h bi = *(const v16h*)(Bg0 + (size_t)kf * 512);
      v16h bf = *(const v16h*)(Bg1 + (size_t)kf * 512);
      v16h bg = *(const v16h*)(Bg2 + (size_t)kf * 512);
      v16h bo = *(const v16h*)(Bg3 + (size_t)kf * 512);
      ci = wmma16(a, bi, ci);
      cf = wmma16(a, bf, cf);
      cg = wmma16(a, bg, cg);
      co = wmma16(a, bo, co);
    }
    float hval[8];
#pragma unroll
    for (int v = 0; v < 8; ++v) {
      float ig = sigm(ci[v]);
      float fg = sigm(cf[v]);
      float gg = tanhf(cg[v]);
      float og = sigm(co[v]);
      c[v] = fg * c[v] + ig * gg;
      hval[v] = og * tanhf(c[v]);
    }
    __syncthreads();                    // all hA reads of this step done
#pragma unroll
    for (int v = 0; v < 8; ++v) {
      hA[baseW + v * 16] = (half_t)hval[v];
      int b = mt * 16 + lrow * 8 + v;
      int tout = (dir == 1) ? rev[b * T_ + t] : t;
      float mk = (word[b * T_ + tout] > 0) ? 1.0f : 0.0f;
      xout[(size_t)(b * T_ + tout) * HID_ + dir * H2_ + jcol] = (half_t)(hval[v] * mk);
    }
    __syncthreads();                    // hA ready for next step
  }
}

// ---- Viterbi decode: one wave per sequence, lane = label --------------------
// emissions are t-major rows: emis[(t*B + b)*32 + l]
__global__ void k_viterbi(const float* __restrict__ emis,
                          const int* __restrict__ word,
                          const float* __restrict__ startT,
                          const float* __restrict__ endT,
                          const float* __restrict__ trans,
                          int* __restrict__ hist,            // [(T-1),B,16]
                          float* __restrict__ outTags) {     // d_out, [B,T]
  int b = blockIdx.x;
  int l = threadIdx.x;
  bool act = l < NT_;
  float score = act ? startT[l] + emis[(size_t)(0 * B_ + b) * 32 + l] : -1e30f;
  float trow[NT_];
#pragma unroll
  for (int lp = 0; lp < NT_; ++lp) trow[lp] = act ? trans[lp * NT_ + l] : 0.0f;

  for (int t = 1; t < T_; ++t) {
    float em = act ? emis[(size_t)(t * B_ + b) * 32 + l] : 0.0f;
    bool m = word[b * T_ + t] > 0;
    float best = -1e30f; int bi = 0;
#pragma unroll
    for (int lp = 0; lp < NT_; ++lp) {
      float s = __shfl(score, lp, 32);
      float cand = s + trow[lp];
      if (cand > best) { best = cand; bi = lp; }
    }
    if (act) hist[((size_t)(t - 1) * B_ + b) * 16 + l] = bi;
    if (m && act) score = best + em;
  }
  score = act ? score + endT[l] : -1e30f;
  int bestl = 0; float bestv = -1e30f;
#pragma unroll
  for (int lp = 0; lp < NT_; ++lp) {
    float s = __shfl(score, lp, 32);
    if (s > bestv) { bestv = s; bestl = lp; }
  }
  if (l == 0) {
    int tag = bestl;
    outTags[b * T_ + (T_ - 1)] = (word[b * T_ + (T_ - 1)] > 0) ? (float)tag : 0.0f;
    for (int t = T_ - 2; t >= 0; --t) {
      bool m = word[b * T_ + t + 1] > 0;
      if (m) tag = hist[((size_t)t * B_ + b) * 16 + tag];
      outTags[b * T_ + t] = (word[b * T_ + t] > 0) ? (float)tag : 0.0f;
    }
  }
}

// ---------------------------------------------------------------------------
extern "C" void kernel_launch(void* const* d_in, const int* in_sizes, int n_in,
                              void* d_out, int out_size, void* d_ws, size_t ws_size,
                              hipStream_t stream) {
  const int*   word  = (const int*)d_in[0];
  const float* embed = (const float*)d_in[1];
  const float* wih0  = (const float*)d_in[2];
  const float* whh0  = (const float*)d_in[3];
  const float* bih0  = (const float*)d_in[4];
  const float* bhh0  = (const float*)d_in[5];
  const float* wih1  = (const float*)d_in[6];
  const float* whh1  = (const float*)d_in[7];
  const float* bih1  = (const float*)d_in[8];
  const float* bhh1  = (const float*)d_in[9];
  const float* Wout  = (const float*)d_in[10];
  const float* bout  = (const float*)d_in[11];
  const float* startT = (const float*)d_in[12];
  const float* endT   = (const float*)d_in[13];
  const float* trans  = (const float*)d_in[14];
  float* out = (float*)d_out;
  (void)in_sizes; (void)n_in; (void)out_size; (void)ws_size;

  char* ws = (char*)d_ws;
  size_t off = 0;
  auto alloc = [&](size_t bytes) -> char* {
    char* p = ws + off;
    off += (bytes + 255) & ~(size_t)255;
    return p;
  };
  int*    rev   = (int*)alloc((size_t)M_ * 4);
  float*  bc0   = (float*)alloc(2048 * 4);
  float*  bc1   = (float*)alloc(2048 * 4);
  float*  bo    = (float*)alloc(32 * 4);
  half_t* Bih0  = (half_t*)alloc((size_t)2 * 64 * 10 * 512 * 2);
  half_t* Bhh0  = (half_t*)alloc((size_t)2 * 64 *  8 * 512 * 2);
  half_t* Bih1  = (half_t*)alloc((size_t)2 * 64 * 16 * 512 * 2);
  half_t* Bhh1  = (half_t*)alloc((size_t)2 * 64 *  8 * 512 * 2);
  half_t* BWo   = (half_t*)alloc((size_t)2 * 16 * 512 * 2);
  half_t* Apack = (half_t*)alloc((size_t)2 * 1024 * 16 * 512 * 2);  // reused l0/l1/emis
  half_t* xg    = (half_t*)alloc((size_t)2 * 1024 * 64 * 256 * 2); // tile layout, reused
  half_t* x1    = (half_t*)alloc((size_t)M_ * 512 * 2);
  half_t* x2    = (half_t*)alloc((size_t)M_ * 512 * 2);
  float*  emis  = (float*)alloc((size_t)M_ * 32 * 4);
  int*    hist  = (int*)alloc((size_t)(T_ - 1) * B_ * 16 * 4);

  // --- prep: lens (-> d_out tail) + reverse indices
  k_prep<<<B_, T_, 0, stream>>>(word, rev, out + M_);

  // --- weight packing
  for (int d = 0; d < 2; ++d) {
    int f0 = 64 * 10;
    k_pack_B<<<(f0 * 32 + 255) / 256, 256, 0, stream>>>(
        wih0 + (size_t)d * 1024 * 300, Bih0 + (size_t)d * 64 * 10 * 512, 1024, 300, 10, f0);
    int f1 = 64 * 8;
    k_pack_B<<<(f1 * 32 + 255) / 256, 256, 0, stream>>>(
        whh0 + (size_t)d * 1024 * 256, Bhh0 + (size_t)d * 64 * 8 * 512, 1024, 256, 8, f1);
    int f2 = 64 * 16;
    k_pack_B<<<(f2 * 32 + 255) / 256, 256, 0, stream>>>(
        wih1 + (size_t)d * 1024 * 512, Bih1 + (size_t)d * 64 * 16 * 512, 1024, 512, 16, f2);
    k_pack_B<<<(f1 * 32 + 255) / 256, 256, 0, stream>>>(
        whh1 + (size_t)d * 1024 * 256, Bhh1 + (size_t)d * 64 * 8 * 512, 1024, 256, 8, f1);
  }
  {
    int fo = 2 * 16;   // Npad=32 -> 2 N-tiles, Kf=16
    k_pack_B<<<(fo * 32 + 255) / 256, 256, 0, stream>>>(Wout, BWo, NT_, 512, 16, fo);
  }
  k_bias<<<8, 256, 0, stream>>>(bih0, bhh0, bih1, bhh1, bout, bc0, bc1, bo);

  // --- layer 0
  for (int d = 0; d < 2; ++d) {
    int nf = 1024 * 10;
    k_pack_A_embed<<<(nf * 32 + 255) / 256, 256, 0, stream>>>(
        word, rev, embed, Apack + (size_t)d * 1024 * 10 * 512, d, nf);
  }
  for (int d = 0; d < 2; ++d) {
    int waves = (1024 / 2) * (64 / 2);
    k_gemm<<<waves * 32 / 256, 256, 0, stream>>>(
        Apack + (size_t)d * 1024 * 10 * 512, Bih0 + (size_t)d * 64 * 10 * 512,
        bc0 + d * 1024, xg + (size_t)d * 1024 * 64 * 256, 64, 10, 0, 2);
  }
  k_lstm<<<2, 1024, 0, stream>>>(xg, Bhh0, word, rev, x1);

  // --- layer 1
  for (int d = 0; d < 2; ++d) {
    int nf = 1024 * 16;
    k_pack_A_h<<<(nf * 32 + 255) / 256, 256, 0, stream>>>(
        x1, rev, Apack + (size_t)d * 1024 * 16 * 512, 512, d, nf);
  }
  for (int d = 0; d < 2; ++d) {
    int waves = (1024 / 2) * (64 / 2);
    k_gemm<<<waves * 32 / 256, 256, 0, stream>>>(
        Apack + (size_t)d * 1024 * 16 * 512, Bih1 + (size_t)d * 64 * 16 * 512,
        bc1 + d * 1024, xg + (size_t)d * 1024 * 64 * 256, 64, 16, 0, 2);
  }
  k_lstm<<<2, 1024, 0, stream>>>(xg, Bhh1, word, rev, x2);

  // --- emissions (N padded to 32, f32 t-major rows) + Viterbi
  {
    int nf = 1024 * 16;
    k_pack_A_h<<<(nf * 32 + 255) / 256, 256, 0, stream>>>(x2, rev, Apack, 512, 0, nf);
    int waves = (1024 / 2) * (2 / 2);
    k_gemm<<<waves * 32 / 256, 256, 0, stream>>>(Apack, BWo, bo, emis, 2, 16, 32, 0);
  }
  k_viterbi<<<B_, 32, 0, stream>>>(emis, word, startT, endT, trans, hist, out);
}
